// GCNAutoencoder_22428319219864
// MI455X (gfx1250) — matrix operand order repaired and checked by reading, not compiled, not run on hardware
//
#include <hip/hip_runtime.h>
#include <hip/hip_bf16.h>
#include <math.h>

// ---------------------------------------------------------------------------
// GCN autoencoder for MI455X (gfx1250, wave32, WMMA).
// Dense GEMMs + Gram matrices via V_WMMA_F32_16X16X32_BF16 with bf16 operands
// resident in memory, f32 accumulate, ping-pong double-buffered fragment
// loads (2x unrolled k-loop). SpMM via f32 global atomics.
// ---------------------------------------------------------------------------

typedef __bf16 bf16_t;
typedef __attribute__((ext_vector_type(16))) __bf16 v16bf;
typedef __attribute__((ext_vector_type(8)))  float  v8f;

#define N_NODE 8192
#define N_EDGE 131072

union FragU { v16bf v; uint4 q[2]; };

__device__ __forceinline__ v16bf make_frag(uint4 lo, uint4 hi) {
    FragU u; u.q[0] = lo; u.q[1] = hi; return u.v;
}

// One k-step worth of raw fragment data (1 A-frag + 4 B-frags, 10 x b128).
struct FragSet {
    uint4 al, ah, b0l, b0h, b1l, b1h, b2l, b2h, b3l, b3h;
};

__device__ __forceinline__ void load_set(FragSet& f,
                                         const bf16_t* aP,  const bf16_t* b0P,
                                         const bf16_t* b1P, const bf16_t* b2P,
                                         const bf16_t* b3P, int k) {
    f.al  = *(const uint4*)(aP  + k);  f.ah  = *(const uint4*)(aP  + k + 8);
    f.b0l = *(const uint4*)(b0P + k);  f.b0h = *(const uint4*)(b0P + k + 8);
    f.b1l = *(const uint4*)(b1P + k);  f.b1h = *(const uint4*)(b1P + k + 8);
    f.b2l = *(const uint4*)(b2P + k);  f.b2h = *(const uint4*)(b2P + k + 8);
    f.b3l = *(const uint4*)(b3P + k);  f.b3h = *(const uint4*)(b3P + k + 8);
}

__device__ __forceinline__ void mma_set(const FragSet& f,
                                        v8f& acc0, v8f& acc1, v8f& acc2, v8f& acc3) {
    v16bf a  = make_frag(f.al,  f.ah);
    v16bf b0 = make_frag(f.b0l, f.b0h);
    v16bf b1 = make_frag(f.b1l, f.b1h);
    v16bf b2 = make_frag(f.b2l, f.b2h);
    v16bf b3 = make_frag(f.b3l, f.b3h);
    acc0 = __builtin_amdgcn_wmma_f32_16x16x32_bf16(false, a, false, b0, (short)0, acc0, false, false);
    acc1 = __builtin_amdgcn_wmma_f32_16x16x32_bf16(false, a, false, b1, (short)0, acc1, false, false);
    acc2 = __builtin_amdgcn_wmma_f32_16x16x32_bf16(false, a, false, b2, (short)0, acc2, false, false);
    acc3 = __builtin_amdgcn_wmma_f32_16x16x32_bf16(false, a, false, b3, (short)0, acc3, false, false);
}

// ---- GEMM: C[M,N] = act(A[M,K] @ BT[N,K]^T) --------------------------------
// A, BT bf16 row-major; C f32. ACT: 0 = identity, 1 = tanh, 2 = sigmoid (+NT).
// One wave owns a 16x64 tile: 1 A-frag shared by 4 B-frags, 4 accumulators.
// K is always a multiple of 64 here, so the k-loop is 2x unrolled with
// ping-pong fragment sets (no live-range overlap -> no register copies).
//
// Fragment layouts used here (contiguous k; A and B share the order):
//   A 16x32: lane m = row m; lanes 0-15 hold K0..15, lanes 16-31 K16..31
//            -> two b128 loads at (k0+half*16) and +8 elems.
//   B 32x16: lane n = col n; lanes 0-15 hold K0..15, lanes 16-31 K16..31
//            -> two b128 loads at (k0+half*16) and +8 elems (BT is [N,K]).

template<int ACT>
__global__ __launch_bounds__(256)
void gemm_wmma_kernel(const bf16_t* __restrict__ A, const bf16_t* __restrict__ BT,
                      float* __restrict__ C, int M, int N, int K)
{
    const int lane = threadIdx.x & 31;
    const int wave = threadIdx.x >> 5;
    const int half = lane >> 4;
    const int l16  = lane & 15;
    const int tilesN = N >> 6;
    const int tile = blockIdx.x * 8 + wave;
    if (tile >= (M >> 4) * tilesN) return;   // uniform per wave
    const int tm = tile / tilesN;
    const int tn = tile % tilesN;

    const bf16_t* aP  = A  + (size_t)(tm * 16 + l16) * K + half * 16;
    const bf16_t* b0P = BT + (size_t)(tn * 64 +  0 + l16) * K + half * 16;
    const bf16_t* b1P = BT + (size_t)(tn * 64 + 16 + l16) * K + half * 16;
    const bf16_t* b2P = BT + (size_t)(tn * 64 + 32 + l16) * K + half * 16;
    const bf16_t* b3P = BT + (size_t)(tn * 64 + 48 + l16) * K + half * 16;

    v8f acc0 = {}; v8f acc1 = {}; v8f acc2 = {}; v8f acc3 = {};

    FragSet s0, s1;
    load_set(s0, aP, b0P, b1P, b2P, b3P, 0);

    for (int k0 = 0; k0 < K; k0 += 64) {
        load_set(s1, aP, b0P, b1P, b2P, b3P, k0 + 32);   // in range: K % 64 == 0
        mma_set(s0, acc0, acc1, acc2, acc3);             // consumes s0
        if (k0 + 64 < K)
            load_set(s0, aP, b0P, b1P, b2P, b3P, k0 + 64);
        mma_set(s1, acc0, acc1, acc2, acc3);             // consumes s1
    }

    // C/D layout: lane holds col l16 (+j*16); VGPR v holds row v (+8 if half).
    const size_t cb = (size_t)(tm * 16 + half * 8) * N + (size_t)tn * 64 + l16;
    #pragma unroll
    for (int v = 0; v < 8; ++v) {
        float y0 = acc0[v], y1 = acc1[v], y2 = acc2[v], y3 = acc3[v];
        if (ACT == 1) {
            y0 = tanhf(y0); y1 = tanhf(y1); y2 = tanhf(y2); y3 = tanhf(y3);
        } else if (ACT == 2) {
            y0 = 1.f / (1.f + __expf(-y0)); y1 = 1.f / (1.f + __expf(-y1));
            y2 = 1.f / (1.f + __expf(-y2)); y3 = 1.f / (1.f + __expf(-y3));
        }
        float* cp = C + cb + (size_t)v * N;
        if (ACT == 2) {   // Gram outputs: 512MB, never re-read -> NT stores
            __builtin_nontemporal_store(y0, cp +  0);
            __builtin_nontemporal_store(y1, cp + 16);
            __builtin_nontemporal_store(y2, cp + 32);
            __builtin_nontemporal_store(y3, cp + 48);
        } else {
            cp[0] = y0; cp[16] = y1; cp[32] = y2; cp[48] = y3;
        }
    }
}

// ---- SpMM: out[row] += val * S[col], per edge, float4 granularity ----------

__global__ __launch_bounds__(256)
void spmm_kernel(const int* __restrict__ rows, const int* __restrict__ cols,
                 const float* __restrict__ vals, const float* __restrict__ S,
                 float* __restrict__ out, int qshift)
{
    const long long idx = (long long)blockIdx.x * blockDim.x + threadIdx.x;
    const int e = (int)(idx >> qshift);
    if (e >= N_EDGE) return;
    const int q = ((int)idx & ((1 << qshift) - 1)) << 2;
    const int D = 4 << qshift;
    const float v = vals[e];
    const float4 s = *(const float4*)(S + (size_t)cols[e] * D + q);
    float* o = out + (size_t)rows[e] * D + q;
    atomicAdd(o + 0, v * s.x);
    atomicAdd(o + 1, v * s.y);
    atomicAdd(o + 2, v * s.z);
    atomicAdd(o + 3, v * s.w);
}

// ---- utility kernels -------------------------------------------------------

__global__ void zero_kernel(float4* __restrict__ p, int n4) {
    int i = blockIdx.x * blockDim.x + threadIdx.x;
    if (i < n4) p[i] = make_float4(0.f, 0.f, 0.f, 0.f);
}

// f32 -> bf16, 8 elements per thread
__global__ void f2bf_kernel(const float* __restrict__ src, bf16_t* __restrict__ dst, int n8) {
    int i = blockIdx.x * blockDim.x + threadIdx.x;
    if (i >= n8) return;
    const float4* s = (const float4*)src + (size_t)i * 2;
    float4 a = s[0], b = s[1];
    union { uint4 q; bf16_t h[8]; } u;
    u.h[0] = (bf16_t)a.x; u.h[1] = (bf16_t)a.y; u.h[2] = (bf16_t)a.z; u.h[3] = (bf16_t)a.w;
    u.h[4] = (bf16_t)b.x; u.h[5] = (bf16_t)b.y; u.h[6] = (bf16_t)b.z; u.h[7] = (bf16_t)b.w;
    *((uint4*)dst + i) = u.q;
}

// W[K,N] f32 -> Wt[N,K] bf16
__global__ void transpose_cvt_kernel(const float* __restrict__ W, bf16_t* __restrict__ Wt,
                                     int K, int N) {
    int i = blockIdx.x * blockDim.x + threadIdx.x;
    if (i < K * N) {
        int k = i / N, n = i - k * N;
        Wt[(size_t)n * K + k] = (bf16_t)W[i];
    }
}

// ---- host side -------------------------------------------------------------

static void launch_gemm(int act, const bf16_t* A, const bf16_t* BT, float* C,
                        int M, int N, int K, hipStream_t s) {
    int tiles = (M / 16) * (N / 64);
    dim3 g((tiles + 7) / 8), b(256);
    if (act == 0)      gemm_wmma_kernel<0><<<g, b, 0, s>>>(A, BT, C, M, N, K);
    else if (act == 1) gemm_wmma_kernel<1><<<g, b, 0, s>>>(A, BT, C, M, N, K);
    else               gemm_wmma_kernel<2><<<g, b, 0, s>>>(A, BT, C, M, N, K);
}

static void launch_spmm(const int* rows, const int* cols, const float* vals,
                        const float* S, float* out, int D, hipStream_t s) {
    int n4 = (N_NODE * D) / 4;
    zero_kernel<<<(n4 + 255) / 256, 256, 0, s>>>((float4*)out, n4);
    int qshift = 31 - __builtin_clz((unsigned)D) - 2;      // log2(D/4)
    long long threads = (long long)N_EDGE << qshift;
    spmm_kernel<<<(unsigned)((threads + 255) / 256), 256, 0, s>>>(rows, cols, vals, S, out, qshift);
}

static void launch_f2bf(const float* src, bf16_t* dst, int n, hipStream_t s) {
    int n8 = n / 8;
    f2bf_kernel<<<(n8 + 255) / 256, 256, 0, s>>>(src, dst, n8);
}

extern "C" void kernel_launch(void* const* d_in, const int* in_sizes, int n_in,
                              void* d_out, int out_size, void* d_ws, size_t ws_size,
                              hipStream_t stream) {
    const float* x    = (const float*)d_in[0];
    const int*   arow = (const int*)  d_in[1];
    const int*   acol = (const int*)  d_in[2];
    const float* aval = (const float*)d_in[3];
    const float* W[6] = { (const float*)d_in[4], (const float*)d_in[5],
                          (const float*)d_in[6], (const float*)d_in[7],
                          (const float*)d_in[8], (const float*)d_in[9] };
    const int wk[6] = {512, 256, 128,  64, 128, 256};   // weight K (in)
    const int wn[6] = {256, 128,  64, 128, 256, 512};   // weight N (out)

    // workspace: bufS(f32) | bufH(f32) | bufB(bf16) | wt[0..5](bf16)
    float*  bufS = (float*)d_ws;                               // 8192*512 f32
    float*  bufH = bufS + (size_t)N_NODE * 512;                // 8192*512 f32
    bf16_t* bufB = (bf16_t*)(bufH + (size_t)N_NODE * 512);     // 8192*512 bf16
    bf16_t* wt[6];
    {
        bf16_t* p = bufB + (size_t)N_NODE * 512;
        for (int i = 0; i < 6; ++i) { wt[i] = p; p += (size_t)wk[i] * wn[i]; }
    }

    // output layout: z_x | z_adj | x_hat | adj_hat
    float* out     = (float*)d_out;
    float* z_x     = out;
    float* z_adj   = out + (size_t)N_NODE * 64;
    float* x_hat   = z_adj + (size_t)N_NODE * N_NODE;
    float* adj_hat = x_hat + (size_t)N_NODE * 512;

    // transpose+convert (tiny) weights so B-fragments are K-contiguous bf16
    for (int i = 0; i < 6; ++i) {
        int n = wk[i] * wn[i];
        transpose_cvt_kernel<<<(n + 255) / 256, 256, 0, stream>>>(W[i], wt[i], wk[i], wn[i]);
    }

    // ---- encoder ----
    launch_f2bf(x, bufB, N_NODE * 512, stream);                    // x -> bf16
    launch_gemm(1, bufB, wt[0], bufS, N_NODE, 256, 512, stream);   // tanh(x@W1)
    launch_spmm(arow, acol, aval, bufS, bufH, 256, stream);        // H1 (f32)
    launch_f2bf(bufH, bufB, N_NODE * 256, stream);
    launch_gemm(1, bufB, wt[1], bufS, N_NODE, 128, 256, stream);   // tanh(H1@W2)
    launch_spmm(arow, acol, aval, bufS, bufH, 128, stream);        // H2
    launch_f2bf(bufH, bufB, N_NODE * 128, stream);
    launch_gemm(0, bufB, wt[2], bufS, N_NODE,  64, 128, stream);   // H2@W3 (no act)
    launch_spmm(arow, acol, aval, bufS, z_x,   64, stream);        // z_x -> d_out

    // z_adj = sigmoid(z_x @ z_x^T); bufB holds bf16(z_x) (BT == A, [N,K] row-major)
    launch_f2bf(z_x, bufB, N_NODE * 64, stream);
    launch_gemm(2, bufB, bufB, z_adj, N_NODE, N_NODE, 64, stream);

    // ---- decoder ----
    launch_gemm(1, bufB, wt[3], bufS, N_NODE, 128,  64, stream);   // tanh(z@W4)
    launch_spmm(arow, acol, aval, bufS, bufH, 128, stream);        // H4
    launch_f2bf(bufH, bufB, N_NODE * 128, stream);
    launch_gemm(1, bufB, wt[4], bufS, N_NODE, 256, 128, stream);   // tanh(H4@W5)
    launch_spmm(arow, acol, aval, bufS, bufH, 256, stream);        // H5
    launch_f2bf(bufH, bufB, N_NODE * 256, stream);
    launch_gemm(1, bufB, wt[5], bufS, N_NODE, 512, 256, stream);   // tanh(H5@W6)
    launch_spmm(arow, acol, aval, bufS, x_hat, 512, stream);       // x_hat -> d_out

    // adj_hat = sigmoid(x_hat @ x_hat^T) — dominant 68.7 GFLOP GEMM
    launch_f2bf(x_hat, bufB, N_NODE * 512, stream);
    launch_gemm(2, bufB, bufB, adj_hat, N_NODE, N_NODE, 512, stream);
}
